// MultiHeadAttention_57638461112372
// MI455X (gfx1250) — matrix operand order repaired and checked by reading, not compiled, not run on hardware
//
#include <hip/hip_runtime.h>
#include <hip/hip_bf16.h>

typedef __bf16 bf16;
typedef __attribute__((ext_vector_type(16))) __bf16 v16bf;
typedef __attribute__((ext_vector_type(8)))  __bf16 v8bf;
typedef __attribute__((ext_vector_type(4)))  __bf16 v4bf;
typedef __attribute__((ext_vector_type(8)))  float  v8f;

#define D_MODEL 1024
#define N_HEAD  16
#define DK      64
#define BATCH   2
#define SEQ     2048
#define BS      (BATCH * SEQ)      // 4096
#define BH      (BATCH * N_HEAD)   // 32

// ---------------------------------------------------------------------------
// WMMA helper: D = A(16x32 bf16) * B(32x16 bf16) + C(16x16 f32)
// ---------------------------------------------------------------------------
__device__ __forceinline__ v8f wmma_bf16(v16bf a, v16bf b, v8f c) {
  return __builtin_amdgcn_wmma_f32_16x16x32_bf16(
      /*neg_a=*/false, a, /*neg_b=*/false, b,
      /*c_mod=*/(short)0, c, /*reuse_a=*/false, /*reuse_b=*/false);
}

// ---------------------------------------------------------------------------
// f32 -> bf16 converting tile loader (ROWS x 32, LDS row stride 32).
// Compile-time trip count; loads batched before converts for MLP.
// ---------------------------------------------------------------------------
template<int ROWS, int NTHR>
__device__ __forceinline__ void tile_f32_ld(bf16* dst, const float* src,
                                            int ld, int tid) {
  constexpr int ITER = (ROWS * 32) / (NTHR * 4);
  float4 f[ITER];
#pragma unroll
  for (int it = 0; it < ITER; ++it) {
    const int idx = tid * 4 + it * NTHR * 4;
    const int r = idx >> 5, c = idx & 31;
    f[it] = *(const float4*)(src + (size_t)r * ld + c);
  }
#pragma unroll
  for (int it = 0; it < ITER; ++it) {
    const int idx = tid * 4 + it * NTHR * 4;
    v4bf q = { (bf16)f[it].x, (bf16)f[it].y, (bf16)f[it].z, (bf16)f[it].w };
    *(v4bf*)(dst + idx) = q;
  }
}

// ---------------------------------------------------------------------------
// CDNA5 async global->LDS copy (ROWS x COLS bf16, LDS row stride COLS).
// GLOBAL_LOAD_ASYNC_TO_LDS_B128: per-lane 16B, tracked with ASYNCcnt.
// LDS offset = low 32 bits of the flat shared pointer (aperture encoding).
// ---------------------------------------------------------------------------
template<int ROWS, int COLS, int NTHR>
__device__ __forceinline__ void tile_bf16_async(bf16* dst, const bf16* src,
                                                int ld, int tid) {
  constexpr int ITER = (ROWS * COLS) / (NTHR * 8);
#pragma unroll
  for (int it = 0; it < ITER; ++it) {
    const int idx = tid * 8 + it * NTHR * 8;
    const int r = idx / COLS, c = idx % COLS;
    const unsigned lds_off = (unsigned)(uintptr_t)(dst + idx);
    const unsigned long long ga =
        (unsigned long long)(uintptr_t)(src + (size_t)r * ld + c);
    asm volatile("global_load_async_to_lds_b128 %0, %1, off"
                 :: "v"(lds_off), "v"(ga) : "memory");
  }
}

__device__ __forceinline__ void wait_async0() {
  asm volatile("s_wait_asynccnt 0x0" ::: "memory");
}

// ---------------------------------------------------------------------------
// Fragment loads from LDS per CDNA5 ISA VGPR layouts (wave32).
// A 16x32: lane m=l&15, hi=l>>4; halves 0..7 -> K=8*hi+h, 8..15 -> K=16+8*hi+(h-8)
// B stored K-contiguous per column (B^T rows): same K order as A.
// ---------------------------------------------------------------------------
__device__ __forceinline__ v16bf frag_a(const bf16* base, int ldl, int lane) {
  const int m = lane & 15, hi = lane >> 4;
  union { v16bf v; v8bf h[2]; } u;
  u.h[0] = *(const v8bf*)(base + m * ldl + 8 * hi);
  u.h[1] = *(const v8bf*)(base + m * ldl + 16 + 8 * hi);
  return u.v;
}

__device__ __forceinline__ v16bf frag_b(const bf16* base, int ldl, int lane) {
  const int n = lane & 15, hi = lane >> 4;
  union { v16bf v; v8bf h[2]; } u;
  u.h[0] = *(const v8bf*)(base + n * ldl + 8 * hi);
  u.h[1] = *(const v8bf*)(base + n * ldl + 16 + 8 * hi);
  return u.v;
}

// ===========================================================================
// Kernel 1: QKV projection.  Y = X @ W^T + b -> head layouts:
//   p=0: Qh[bh][s][d]  p=1: Kh[bh][s][d]  p=2: VhT[bh][d][s]   (bf16)
// ===========================================================================
__global__ __launch_bounds__(128)
void qkv_proj_kernel(const float* __restrict__ q, const float* __restrict__ k,
                     const float* __restrict__ v,
                     const float* __restrict__ wq, const float* __restrict__ wk,
                     const float* __restrict__ wv,
                     const float* __restrict__ bq, const float* __restrict__ bk,
                     const float* __restrict__ bv,
                     bf16* __restrict__ Qh, bf16* __restrict__ Kh,
                     bf16* __restrict__ VhT) {
  __shared__ __align__(16) bf16 As[64 * 32];
  __shared__ __align__(16) bf16 Bs[64 * 32];

  const int p = blockIdx.z;
  const float* X    = (p == 0) ? q  : (p == 1) ? k  : v;
  const float* W    = (p == 0) ? wq : (p == 1) ? wk : wv;
  const float* bias = (p == 0) ? bq : (p == 1) ? bk : bv;

  const int mBase = blockIdx.y * 64;
  const int nBase = blockIdx.x * 64;
  const int tid = threadIdx.x, lane = tid & 31, wave = tid >> 5;
  const int m0 = (wave >> 1) * 32, n0 = (wave & 1) * 32;

  v8f acc[2][2] = {};
  for (int kt = 0; kt < D_MODEL; kt += 32) {
    __syncthreads();
    tile_f32_ld<64, 128>(As, X + (size_t)mBase * D_MODEL + kt, D_MODEL, tid);
    tile_f32_ld<64, 128>(Bs, W + (size_t)nBase * D_MODEL + kt, D_MODEL, tid);
    __syncthreads();
    const v16bf a0 = frag_a(As + (m0) * 32, 32, lane);
    const v16bf a1 = frag_a(As + (m0 + 16) * 32, 32, lane);
    const v16bf b0 = frag_b(Bs + (n0) * 32, 32, lane);
    const v16bf b1 = frag_b(Bs + (n0 + 16) * 32, 32, lane);
    acc[0][0] = wmma_bf16(a0, b0, acc[0][0]);
    acc[0][1] = wmma_bf16(a0, b1, acc[0][1]);
    acc[1][0] = wmma_bf16(a1, b0, acc[1][0]);
    acc[1][1] = wmma_bf16(a1, b1, acc[1][1]);
  }

  const int hi = lane >> 4, col = lane & 15;
#pragma unroll
  for (int fm = 0; fm < 2; ++fm)
#pragma unroll
    for (int fn = 0; fn < 2; ++fn)
#pragma unroll
      for (int i = 0; i < 8; ++i) {
        const int m = mBase + m0 + fm * 16 + i + 8 * hi;
        const int n = nBase + n0 + fn * 16 + col;
        const int b = m >> 11, s = m & (SEQ - 1);
        const int h = n >> 6,  d = n & (DK - 1);
        const int bh = b * N_HEAD + h;
        const bf16 o = (bf16)(acc[fm][fn][i] + bias[n]);
        if (p == 0)      Qh[((size_t)bh * SEQ + s) * DK + d] = o;
        else if (p == 1) Kh[((size_t)bh * SEQ + s) * DK + d] = o;
        else             VhT[((size_t)bh * DK + d) * SEQ + s] = o;
      }
}

// ===========================================================================
// Kernel 2: scores[bh][s][t] = (Q . K) / 8.  Whole K=64 loaded once via
// async-to-LDS copies; 2 K-steps, 8 WMMAs, no per-step barriers.
// ===========================================================================
__global__ __launch_bounds__(128)
void scores_kernel(const bf16* __restrict__ Qh, const bf16* __restrict__ Kh,
                   float* __restrict__ attn) {
  __shared__ __align__(16) bf16 As[64 * 64];
  __shared__ __align__(16) bf16 Bs[64 * 64];

  const int bh = blockIdx.z;
  const int mBase = blockIdx.y * 64;
  const int nBase = blockIdx.x * 64;
  const bf16* Q = Qh + (size_t)bh * SEQ * DK;
  const bf16* K = Kh + (size_t)bh * SEQ * DK;
  float* S = attn + (size_t)bh * SEQ * SEQ;

  const int tid = threadIdx.x, lane = tid & 31, wave = tid >> 5;
  const int m0 = (wave >> 1) * 32, n0 = (wave & 1) * 32;

  tile_bf16_async<64, 64, 128>(As, Q + (size_t)mBase * DK, DK, tid);
  tile_bf16_async<64, 64, 128>(Bs, K + (size_t)nBase * DK, DK, tid);
  wait_async0();
  __syncthreads();

  v8f acc[2][2] = {};
#pragma unroll
  for (int kt = 0; kt < DK; kt += 32) {
    const v16bf a0 = frag_a(As + (m0) * 64 + kt, 64, lane);
    const v16bf a1 = frag_a(As + (m0 + 16) * 64 + kt, 64, lane);
    const v16bf b0 = frag_b(Bs + (n0) * 64 + kt, 64, lane);
    const v16bf b1 = frag_b(Bs + (n0 + 16) * 64 + kt, 64, lane);
    acc[0][0] = wmma_bf16(a0, b0, acc[0][0]);
    acc[0][1] = wmma_bf16(a0, b1, acc[0][1]);
    acc[1][0] = wmma_bf16(a1, b0, acc[1][0]);
    acc[1][1] = wmma_bf16(a1, b1, acc[1][1]);
  }

  const int hi = lane >> 4, col = lane & 15;
  const float scale = 0.125f;  // 1/sqrt(64)
#pragma unroll
  for (int fm = 0; fm < 2; ++fm)
#pragma unroll
    for (int fn = 0; fn < 2; ++fn)
#pragma unroll
      for (int i = 0; i < 8; ++i) {
        const int m = mBase + m0 + fm * 16 + i + 8 * hi;
        const int n = nBase + n0 + fn * 16 + col;
        S[(size_t)m * SEQ + n] = acc[fm][fn][i] * scale;
      }
}

// ===========================================================================
// Kernel 3: in-place row softmax, one 256-thread block per row of 2048.
// ===========================================================================
__global__ __launch_bounds__(256)
void softmax_kernel(float* __restrict__ attn) {
  __shared__ float red[256];
  float* row = attn + (size_t)blockIdx.x * SEQ;
  const int tid = threadIdx.x;

  float vals[8];
  float m = -3.4e38f;
#pragma unroll
  for (int i = 0; i < 8; ++i) {
    vals[i] = row[tid + i * 256];
    m = fmaxf(m, vals[i]);
  }
  red[tid] = m;
  __syncthreads();
  for (int s = 128; s > 0; s >>= 1) {
    if (tid < s) red[tid] = fmaxf(red[tid], red[tid + s]);
    __syncthreads();
  }
  m = red[0];
  __syncthreads();

  float sum = 0.0f;
#pragma unroll
  for (int i = 0; i < 8; ++i) {
    vals[i] = __expf(vals[i] - m);
    sum += vals[i];
  }
  red[tid] = sum;
  __syncthreads();
  for (int s = 128; s > 0; s >>= 1) {
    if (tid < s) red[tid] += red[tid + s];
    __syncthreads();
  }
  const float inv = 1.0f / red[0];
#pragma unroll
  for (int i = 0; i < 8; ++i) row[tid + i * 256] = vals[i] * inv;
}

// ===========================================================================
// Kernel 4: out_h = attn @ Vh per (b,h).  M=SEQ, N=64, K=SEQ.
// B tile (VhT) via async copy, overlapped with A tile f32->bf16 conversion.
// ===========================================================================
__global__ __launch_bounds__(128)
void attnv_kernel(const float* __restrict__ attn, const bf16* __restrict__ VhT,
                  bf16* __restrict__ om) {
  __shared__ __align__(16) bf16 As[64 * 32];
  __shared__ __align__(16) bf16 Bs[64 * 32];

  const int bh = blockIdx.z;
  const int mBase = blockIdx.y * 64;
  const float* A = attn + (size_t)bh * SEQ * SEQ;
  const bf16* V = VhT + (size_t)bh * DK * SEQ;

  const int tid = threadIdx.x, lane = tid & 31, wave = tid >> 5;
  const int m0 = (wave >> 1) * 32, n0 = (wave & 1) * 32;

  v8f acc[2][2] = {};
  for (int kt = 0; kt < SEQ; kt += 32) {
    __syncthreads();
    tile_bf16_async<64, 32, 128>(Bs, V + kt, SEQ, tid);       // rows=d, K=t
    tile_f32_ld<64, 128>(As, A + (size_t)mBase * SEQ + kt, SEQ, tid);
    wait_async0();
    __syncthreads();
    const v16bf a0 = frag_a(As + (m0) * 32, 32, lane);
    const v16bf a1 = frag_a(As + (m0 + 16) * 32, 32, lane);
    const v16bf b0 = frag_b(Bs + (n0) * 32, 32, lane);
    const v16bf b1 = frag_b(Bs + (n0 + 16) * 32, 32, lane);
    acc[0][0] = wmma_bf16(a0, b0, acc[0][0]);
    acc[0][1] = wmma_bf16(a0, b1, acc[0][1]);
    acc[1][0] = wmma_bf16(a1, b0, acc[1][0]);
    acc[1][1] = wmma_bf16(a1, b1, acc[1][1]);
  }

  const int b = bh >> 4, h = bh & (N_HEAD - 1);
  const int hi = lane >> 4, col = lane & 15;
#pragma unroll
  for (int fm = 0; fm < 2; ++fm)
#pragma unroll
    for (int fn = 0; fn < 2; ++fn)
#pragma unroll
      for (int i = 0; i < 8; ++i) {
        const int s = mBase + m0 + fm * 16 + i + 8 * hi;
        const int d = n0 + fn * 16 + col;
        om[((size_t)(b * SEQ + s)) * D_MODEL + h * DK + d] = (bf16)acc[fm][fn][i];
      }
}

// ===========================================================================
// Kernel 5: out = om @ w_o^T + b_o  (f32 to d_out).  A (bf16) via async copy.
// ===========================================================================
__global__ __launch_bounds__(128)
void oproj_kernel(const bf16* __restrict__ om, const float* __restrict__ wo,
                  const float* __restrict__ bo, float* __restrict__ out) {
  __shared__ __align__(16) bf16 As[64 * 32];
  __shared__ __align__(16) bf16 Bs[64 * 32];

  const int mBase = blockIdx.y * 64;
  const int nBase = blockIdx.x * 64;
  const int tid = threadIdx.x, lane = tid & 31, wave = tid >> 5;
  const int m0 = (wave >> 1) * 32, n0 = (wave & 1) * 32;

  v8f acc[2][2] = {};
  for (int kt = 0; kt < D_MODEL; kt += 32) {
    __syncthreads();
    tile_bf16_async<64, 32, 128>(As, om + (size_t)mBase * D_MODEL + kt,
                                 D_MODEL, tid);
    tile_f32_ld<64, 128>(Bs, wo + (size_t)nBase * D_MODEL + kt, D_MODEL, tid);
    wait_async0();
    __syncthreads();
    const v16bf a0 = frag_a(As + (m0) * 32, 32, lane);
    const v16bf a1 = frag_a(As + (m0 + 16) * 32, 32, lane);
    const v16bf b0 = frag_b(Bs + (n0) * 32, 32, lane);
    const v16bf b1 = frag_b(Bs + (n0 + 16) * 32, 32, lane);
    acc[0][0] = wmma_bf16(a0, b0, acc[0][0]);
    acc[0][1] = wmma_bf16(a0, b1, acc[0][1]);
    acc[1][0] = wmma_bf16(a1, b0, acc[1][0]);
    acc[1][1] = wmma_bf16(a1, b1, acc[1][1]);
  }

  const int hi = lane >> 4, col = lane & 15;
#pragma unroll
  for (int fm = 0; fm < 2; ++fm)
#pragma unroll
    for (int fn = 0; fn < 2; ++fn)
#pragma unroll
      for (int i = 0; i < 8; ++i) {
        const int m = mBase + m0 + fm * 16 + i + 8 * hi;
        const int n = nBase + n0 + fn * 16 + col;
        out[(size_t)m * D_MODEL + n] = acc[fm][fn][i] + bo[n];
      }
}

// ===========================================================================
// Launch
// ===========================================================================
extern "C" void kernel_launch(void* const* d_in, const int* in_sizes, int n_in,
                              void* d_out, int out_size, void* d_ws, size_t ws_size,
                              hipStream_t stream) {
  (void)in_sizes; (void)n_in; (void)out_size; (void)ws_size;

  const float* q  = (const float*)d_in[0];
  const float* k  = (const float*)d_in[1];
  const float* v  = (const float*)d_in[2];
  const float* wq = (const float*)d_in[3];
  const float* bq = (const float*)d_in[4];
  const float* wk = (const float*)d_in[5];
  const float* bk = (const float*)d_in[6];
  const float* wv = (const float*)d_in[7];
  const float* bv = (const float*)d_in[8];
  const float* wo = (const float*)d_in[9];
  const float* bo = (const float*)d_in[10];

  float* out  = (float*)d_out;                       // [B,S,D]
  float* attn = out + (size_t)BS * D_MODEL;          // [B,H,S,S]

  bf16* Qh  = (bf16*)d_ws;
  bf16* Kh  = Qh  + (size_t)BH * SEQ * DK;
  bf16* VhT = Kh  + (size_t)BH * SEQ * DK;
  bf16* om  = VhT + (size_t)BH * SEQ * DK;

  qkv_proj_kernel<<<dim3(D_MODEL / 64, BS / 64, 3), 128, 0, stream>>>(
      q, k, v, wq, wk, wv, bq, bk, bv, Qh, Kh, VhT);
  scores_kernel<<<dim3(SEQ / 64, SEQ / 64, BH), 128, 0, stream>>>(Qh, Kh, attn);
  softmax_kernel<<<dim3(BH * SEQ), 256, 0, stream>>>(attn);
  attnv_kernel<<<dim3(1, SEQ / 64, BH), 128, 0, stream>>>(attn, VhT, om);
  oproj_kernel<<<dim3(D_MODEL / 64, BS / 64, 1), 128, 0, stream>>>(om, wo, bo, out);
}
